// DotProductAttention_87024627352354
// MI455X (gfx1250) — compile-verified, matrix operand order checked by reading, not run on hardware
//
#include <hip/hip_runtime.h>

// ---- CDNA5 (gfx1250) flash-attention, bf16 WMMA, fp32 accum ----
// exp2-domain online softmax, native bf16 conversions, B-fragment preloading
// so ds_load latency overlaps WMMA issue (staggered s_wait_dscnt).

typedef __attribute__((ext_vector_type(16))) __bf16        v16bf;
typedef __attribute__((ext_vector_type(8)))  __bf16        v8bf;
typedef __attribute__((ext_vector_type(4)))  __bf16        v4bf;
typedef __attribute__((ext_vector_type(8)))  float          v8f;
typedef __attribute__((ext_vector_type(4)))  float          v4f;

#define TQ 2048
#define TKK 2048
#define DD 128
#define BM 128            // queries per workgroup (8 waves x 16 rows)
#define BN 64             // keys per iteration
#define KP 136            // K LDS pitch in bf16 elems (272B rows, 16B aligned, conflict-free)
#define VP 72             // Vt LDS pitch (144B rows)
#define PP 72             // P  LDS pitch
// log2(e)-scaled constants: scores kept in log2 domain so raw v_exp_f32 applies
#define L2E 1.4426950408889634f
#define SCALE_L2E (0.088388347648318447f * L2E)   // (1/sqrt(128)) * log2(e)
#define MASKV_L2E (-1000000.0f * L2E)

union frag16 { v16bf w; v8bf h[2]; };

__global__ __launch_bounds__(256)
void DotProductAttention_fa_kernel(const float* __restrict__ Q,
                                   const float* __restrict__ K,
                                   const float* __restrict__ V,
                                   const long long* __restrict__ VL,
                                   float* __restrict__ O) {
  __shared__ __align__(16) __bf16 Ks[BN * KP];   // K block, row-major (key x dim)
  __shared__ __align__(16) __bf16 Vt[DD * VP];   // V block, transposed (dim x key)
  __shared__ __align__(16) __bf16 Ps[BM * PP];   // P staging (query x key)
  __shared__ int vls[BM];

  const int tid  = threadIdx.x;
  const int lane = tid & 31;
  const int wv   = tid >> 5;       // wave id 0..7
  const int lh   = lane >> 4;      // lane half (0/1) per WMMA layout
  const int lc   = lane & 15;      // column/row-in-half index
  const int b    = blockIdx.y;
  const int q0   = blockIdx.x * BM;

  // ---- per-row valid lengths (int64 -> clamped int) ----
  if (tid < BM) {
    long long vl = VL[(long long)b * TQ + q0 + tid];
    vls[tid] = (int)(vl < 0 ? 0 : (vl > TKK ? (long long)TKK : vl));
  }

  // ---- Q A-fragments: 16x32 bf16 per k-step, 4 steps cover d=128 ----
  // A layout: lanes 0-15 hold row M=lc with K {0..7,16..23}; lanes 16-31 K {8..15,24..31}
  const float* qp = Q + ((long long)b * TQ + q0 + wv * 16 + lc) * DD;
  v16bf qa[4];
#pragma unroll
  for (int s = 0; s < 4; ++s) {
    const int k0 = s * 32 + (lh ? 8 : 0);
    const int k1 = s * 32 + (lh ? 24 : 16);
    v4f f0 = *(const v4f*)(qp + k0);
    v4f f1 = *(const v4f*)(qp + k0 + 4);
    v4f f2 = *(const v4f*)(qp + k1);
    v4f f3 = *(const v4f*)(qp + k1 + 4);
    v16bf u;
#pragma unroll
    for (int e = 0; e < 4; ++e) {
      u[e]      = (__bf16)f0[e];
      u[e + 4]  = (__bf16)f1[e];
      u[e + 8]  = (__bf16)f2[e];
      u[e + 12] = (__bf16)f3[e];
    }
    qa[s] = u;
  }

  __syncthreads();
  int vlj[8];
#pragma unroll
  for (int j = 0; j < 8; ++j) vlj[j] = vls[wv * 16 + j + 8 * lh];

  // ---- running flash-attention state (log2 domain) ----
  v8f o[8];                       // O tile: 16 rows x 128 dims, fp32
  v8f zinit = {};
#pragma unroll
  for (int t = 0; t < 8; ++t) o[t] = zinit;
  float rm[8], rs[8];
#pragma unroll
  for (int j = 0; j < 8; ++j) { rm[j] = -3.0e38f; rs[j] = 0.f; }

  for (int kb = 0; kb < TKK / BN; ++kb) {
    __syncthreads();   // previous iteration's LDS reads done

    // ---- stage K (row-major) and V (transposed) tiles, fp32 -> bf16 ----
#pragma unroll
    for (int i = 0; i < 8; ++i) {
      int idx = tid + i * 256;         // 0..2047 -> 64 rows x 32 float4
      int row = idx >> 5;
      int c4  = (idx & 31) * 4;
      long long gbase = ((long long)b * TKK + (long long)kb * BN + row) * DD + c4;
      v4f f = *(const v4f*)(K + gbase);
      v4bf u4 = { (__bf16)f[0], (__bf16)f[1], (__bf16)f[2], (__bf16)f[3] };
      *(v4bf*)&Ks[row * KP + c4] = u4;
      v4f g = *(const v4f*)(V + gbase);
      Vt[(c4 + 0) * VP + row] = (__bf16)g[0];
      Vt[(c4 + 1) * VP + row] = (__bf16)g[1];
      Vt[(c4 + 2) * VP + row] = (__bf16)g[2];
      Vt[(c4 + 3) * VP + row] = (__bf16)g[3];
    }
    if (kb + 1 < TKK / BN) {  // L2 prefetch of next K tile (global_prefetch_b8)
      __builtin_prefetch(K + ((long long)b * TKK + (long long)(kb + 1) * BN + (tid >> 2)) * DD
                           + (tid & 3) * 32, 0, 1);
    }
    __syncthreads();

    // ---- S = Q * K^T : 4 col-tiles; per tile preload 4 B-fragments then 4 WMMAs ----
    v8f acc[4];
#pragma unroll
    for (int t = 0; t < 4; ++t) {
      const __bf16* kp = &Ks[(t * 16 + lc) * KP];
      frag16 bk[4];
#pragma unroll
      for (int s = 0; s < 4; ++s) {
        bk[s].h[0] = *(const v8bf*)(kp + s * 32 + (lh ? 8 : 0));
        bk[s].h[1] = *(const v8bf*)(kp + s * 32 + (lh ? 24 : 16));
      }
      v8f z = {};
#pragma unroll
      for (int s = 0; s < 4; ++s)
        z = __builtin_amdgcn_wmma_f32_16x16x32_bf16(false, qa[s], false, bk[s].w,
                                                    (short)0, z, false, false);
      acc[t] = z;
    }

    // ---- masked online softmax in log2 domain ----
    const int colbase = kb * BN;
#pragma unroll
    for (int j = 0; j < 8; ++j) {
      float pm = -3.0e38f;
#pragma unroll
      for (int t = 0; t < 4; ++t) {
        int col = colbase + t * 16 + lc;
        float sv = (col < vlj[j]) ? acc[t][j] * SCALE_L2E : MASKV_L2E;
        acc[t][j] = sv;
        pm = fmaxf(pm, sv);
      }
#pragma unroll
      for (int m = 1; m <= 8; m <<= 1) pm = fmaxf(pm, __shfl_xor(pm, m, 32));
      float nm   = fmaxf(rm[j], pm);
      float corr = __builtin_amdgcn_exp2f(rm[j] - nm);   // raw v_exp_f32
      rm[j] = nm;
      float ls = 0.f;
#pragma unroll
      for (int t = 0; t < 4; ++t) {
        float p = __builtin_amdgcn_exp2f(acc[t][j] - nm);
        acc[t][j] = p;
        ls += p;
      }
#pragma unroll
      for (int m = 1; m <= 8; m <<= 1) ls += __shfl_xor(ls, m, 32);
      rs[j] = rs[j] * corr + ls;
#pragma unroll
      for (int t = 0; t < 8; ++t) o[t][j] *= corr;
    }

    // ---- stage P as bf16 in wave-private LDS (C-layout -> A-layout bounce) ----
    const int pr0 = wv * 16;
#pragma unroll
    for (int t = 0; t < 4; ++t)
#pragma unroll
      for (int j = 0; j < 8; ++j)
        Ps[(pr0 + j + 8 * lh) * PP + t * 16 + lc] = (__bf16)acc[t][j];

    // ---- O += P * V : 2 k-steps; per step preload 4 V-fragments per WMMA quad ----
#pragma unroll
    for (int s = 0; s < 2; ++s) {
      const int k0 = s * 32 + (lh ? 8 : 0);
      const int k1 = s * 32 + (lh ? 24 : 16);
      const __bf16* pp = &Ps[(pr0 + lc) * PP];
      frag16 pa;
      pa.h[0] = *(const v8bf*)(pp + k0);
      pa.h[1] = *(const v8bf*)(pp + k1);
#pragma unroll
      for (int t4 = 0; t4 < 8; t4 += 4) {
        frag16 vb[4];
#pragma unroll
        for (int u = 0; u < 4; ++u) {
          const __bf16* vp = &Vt[((t4 + u) * 16 + lc) * VP];
          vb[u].h[0] = *(const v8bf*)(vp + k0);
          vb[u].h[1] = *(const v8bf*)(vp + k1);
        }
#pragma unroll
        for (int u = 0; u < 4; ++u)
          o[t4 + u] = __builtin_amdgcn_wmma_f32_16x16x32_bf16(false, pa.w, false, vb[u].w,
                                                              (short)0, o[t4 + u],
                                                              false, false);
      }
    }
  }

  // ---- normalize and store O (fp32) ----
  float* op = O + ((long long)b * TQ + q0 + wv * 16) * DD;
#pragma unroll
  for (int j = 0; j < 8; ++j) {
    float inv = 1.f / rs[j];
    int mrow = j + 8 * lh;
#pragma unroll
    for (int t = 0; t < 8; ++t)
      op[(long long)mrow * DD + t * 16 + lc] = o[t][j] * inv;
  }
}

extern "C" void kernel_launch(void* const* d_in, const int* in_sizes, int n_in,
                              void* d_out, int out_size, void* d_ws, size_t ws_size,
                              hipStream_t stream) {
  const float*     q  = (const float*)d_in[0];
  const float*     k  = (const float*)d_in[1];
  const float*     v  = (const float*)d_in[2];
  const long long* vl = (const long long*)d_in[3];
  float* out = (float*)d_out;
  const int B = in_sizes[0] / (TQ * DD);
  dim3 grid(TQ / BM, B);
  hipLaunchKernelGGL(DotProductAttention_fa_kernel, grid, dim3(256), 0, stream,
                     q, k, v, vl, out);
}
